// LongBlock_53532472377660
// MI455X (gfx1250) — matrix operand order repaired and checked by reading, not compiled, not run on hardware
//
#include <hip/hip_runtime.h>
#include <hip/hip_bf16.h>
#include <cstdint>
#include <cstddef>

// ---------------------------------------------------------------------------
// LongBlock for MI455X (gfx1250). All heavy math runs through
// v_wmma_f32_16x16x32_bf16 (bf16 in, f32 accumulate). Weights are cast+
// transposed once so both GEMM operands stage with 16B vector copies; the
// GEMM main loop is double-buffered in LDS using gfx1250 async DMA
// (global_load_async_to_lds_b128 + s_wait_asynccnt) when available.
// ---------------------------------------------------------------------------

#define DM      2048      // d_model
#define NH      16        // heads
#define HD      128       // head dim
#define FF      8192
#define KCONV   4
#define NB      2         // batch
#define TSEQ    4096
#define MTOK    (NB*TSEQ) // 8192 tokens
#define EPSF    1e-5f

#define LC      128       // scan chunk length
#define NCH     (TSEQ/LC) // 32 chunks

typedef __bf16 bf16_t;
typedef bf16_t bf16x8  __attribute__((ext_vector_type(8)));
typedef bf16_t bf16x16 __attribute__((ext_vector_type(16)));
typedef float  f32x8   __attribute__((ext_vector_type(8)));

#if __has_builtin(__builtin_amdgcn_global_load_async_to_lds_b128) && \
    __has_builtin(__builtin_amdgcn_s_wait_asynccnt)
#define USE_ASYNC_COPY 1
#define AS1 __attribute__((address_space(1)))
#define AS3 __attribute__((address_space(3)))
// builtin signature (from clang diagnostic): (v4i AS1*, v4i AS3*, Imm offset, Imm cpol)
typedef int v4i_ __attribute__((vector_size(4 * sizeof(int))));
#endif

// ---------------------------------------------------------------------------
// small helpers
// ---------------------------------------------------------------------------
__device__ __forceinline__ float sigmoidf_(float x) { return 1.f / (1.f + __expf(-x)); }
__device__ __forceinline__ float siluf_(float x)    { return x * sigmoidf_(x); }

__device__ __forceinline__ float blockReduce128(float v, float* red) {
    int d = threadIdx.x;
    red[d] = v; __syncthreads();
    #pragma unroll
    for (int off = 64; off > 0; off >>= 1) {
        if (d < off) red[d] += red[d + off];
        __syncthreads();
    }
    float r = red[0]; __syncthreads();
    return r;
}

// ---------------------------------------------------------------------------
// weight cast f32[K,N] -> bf16 transposed [N,K]  (tiled via LDS, once/launch)
// block = 256 (32x8), grid = (N/32, K/32)
// ---------------------------------------------------------------------------
__global__ void transpose_cast_kernel(const float* __restrict__ in, bf16_t* __restrict__ out,
                                      int K, int N) {
    __shared__ bf16_t tile[32][33];
    const int kt = blockIdx.y * 32, nt = blockIdx.x * 32;
    const int tx = threadIdx.x & 31, ty = threadIdx.x >> 5;   // ty: 0..7
    #pragma unroll
    for (int i = 0; i < 32; i += 8)
        tile[ty + i][tx] = (bf16_t)in[(size_t)(kt + ty + i) * N + nt + tx];
    __syncthreads();
    #pragma unroll
    for (int i = 0; i < 32; i += 8)
        out[(size_t)(nt + ty + i) * K + kt + tx] = tile[tx][ty + i];
}

// ---------------------------------------------------------------------------
// LayerNorm over C=2048 per token; optional f32 copy + bf16 copy
// ---------------------------------------------------------------------------
__global__ void ln_kernel(const float* __restrict__ x,
                          const float* __restrict__ w, const float* __restrict__ b,
                          float* __restrict__ of32, bf16_t* __restrict__ obf) {
    const size_t tok = blockIdx.x;
    const float* xr = x + tok * DM;
    float v[8], s = 0.f, ss = 0.f;
    #pragma unroll
    for (int i = 0; i < 8; ++i) {
        float t = xr[threadIdx.x + i * 256];
        v[i] = t; s += t; ss += t * t;
    }
    __shared__ float r1[256], r2[256];
    r1[threadIdx.x] = s; r2[threadIdx.x] = ss;
    __syncthreads();
    #pragma unroll
    for (int off = 128; off > 0; off >>= 1) {
        if (threadIdx.x < off) { r1[threadIdx.x] += r1[threadIdx.x + off];
                                 r2[threadIdx.x] += r2[threadIdx.x + off]; }
        __syncthreads();
    }
    float mean = r1[0] * (1.f / DM);
    float var  = r2[0] * (1.f / DM) - mean * mean;
    float inv  = rsqrtf(var + EPSF);
    #pragma unroll
    for (int i = 0; i < 8; ++i) {
        int c = threadIdx.x + i * 256;
        float hn = (v[i] - mean) * inv * w[c] + b[c];
        if (of32) of32[tok * DM + c] = hn;
        obf[tok * DM + c] = (bf16_t)hn;
    }
}

// ---------------------------------------------------------------------------
// depthwise causal conv (K=4) + SiLU -> bf16
// ---------------------------------------------------------------------------
__global__ void conv_silu_kernel(const float* __restrict__ h,
                                 const float* __restrict__ w,   // [C,1,K] flat
                                 const float* __restrict__ bias,
                                 bf16_t* __restrict__ out) {
    size_t idx = (size_t)blockIdx.x * 256 + threadIdx.x;   // token*DM + c
    int c = (int)(idx % DM);
    size_t tok = idx / DM;
    int t = (int)(tok % TSEQ);
    size_t brow = tok - (size_t)t;                         // b*TSEQ
    float acc = bias[c];
    #pragma unroll
    for (int j = 0; j < KCONV; ++j) {
        int tt = t - (KCONV - 1) + j;
        if (tt >= 0) acc += h[(brow + tt) * DM + c] * w[c * KCONV + j];
    }
    out[idx] = (bf16_t)siluf_(acc);
}

// ---------------------------------------------------------------------------
// WMMA GEMM:  C[M,N] (+epilogue) = A_bf16[M,K] @ BT_bf16[N,K]^T
// block tile 128x128x32, 256 threads = 8 waves (2x4), wave tile 64x32
// double-buffered LDS; async global->LDS DMA when available.
// MODE 0: f32 = acc
// MODE 1: f32 = acc + Res
// MODE 3: bf16 = silu(acc)
// MODE 4: bf16 = acc * bf16_existing (in-place multiply)
// MODE 5: bf16 = sigmoid(acc + bias[col])
// ---------------------------------------------------------------------------
#define GBM 128
#define GBN 128
#define GBK 32
#define LDK 40   // padded LDS k-stride (bf16 elems) -> 80B rows, 16B aligned

template<int MODE>
__global__ __launch_bounds__(256)
void gemm_bf16_kernel(const bf16_t* __restrict__ A, const bf16_t* __restrict__ BT,
                      float* __restrict__ Cf, bf16_t* __restrict__ Cb,
                      const float* __restrict__ Res, const float* __restrict__ bias,
                      int M, int N, int K) {
    __shared__ bf16_t sA[2][GBM * LDK];
    __shared__ bf16_t sB[2][GBN * LDK];   // [n][k]

    const int tid   = threadIdx.x;
    const int wave  = tid >> 5;
    const int lane  = tid & 31;
    const int waveM = wave >> 2;       // 0..1
    const int waveN = wave & 3;        // 0..3
    const int rowBase = blockIdx.y * GBM;
    const int colBase = blockIdx.x * GBN;

    const int lrow = lane & 15;
    const int lhi  = lane >> 4;        // 0/1 selects K-half per ISA layout

    f32x8 acc[4][2] = {};

    // staging: 2 threads per row (A) / per row of BT (B); 16B each
    const int sRow = tid >> 1;
    const int sK   = (tid & 1) * 16;
    const bf16_t* aGlob = A  + (size_t)(rowBase + sRow) * K + sK;
    const bf16_t* bGlob = BT + (size_t)(colBase + sRow) * K + sK;
    const int sOff = sRow * LDK + sK;

#if defined(USE_ASYNC_COPY)
    auto stageTile = [&](int k0, int buf) {
        __builtin_amdgcn_global_load_async_to_lds_b128(
            (AS1 v4i_*)(aGlob + k0),     (AS3 v4i_*)&sA[buf][sOff],     0, 0);
        __builtin_amdgcn_global_load_async_to_lds_b128(
            (AS1 v4i_*)(aGlob + k0 + 8), (AS3 v4i_*)&sA[buf][sOff + 8], 0, 0);
        __builtin_amdgcn_global_load_async_to_lds_b128(
            (AS1 v4i_*)(bGlob + k0),     (AS3 v4i_*)&sB[buf][sOff],     0, 0);
        __builtin_amdgcn_global_load_async_to_lds_b128(
            (AS1 v4i_*)(bGlob + k0 + 8), (AS3 v4i_*)&sB[buf][sOff + 8], 0, 0);
    };
#else
    bf16x8 ra0, ra1, rb0, rb1;
    auto loadTile = [&](int k0) {
        ra0 = *reinterpret_cast<const bf16x8*>(aGlob + k0);
        ra1 = *reinterpret_cast<const bf16x8*>(aGlob + k0 + 8);
        rb0 = *reinterpret_cast<const bf16x8*>(bGlob + k0);
        rb1 = *reinterpret_cast<const bf16x8*>(bGlob + k0 + 8);
    };
    auto storeTile = [&](int buf) {
        *reinterpret_cast<bf16x8*>(&sA[buf][sOff])     = ra0;
        *reinterpret_cast<bf16x8*>(&sA[buf][sOff + 8]) = ra1;
        *reinterpret_cast<bf16x8*>(&sB[buf][sOff])     = rb0;
        *reinterpret_cast<bf16x8*>(&sB[buf][sOff + 8]) = rb1;
    };
#endif

    const int nk = K / GBK;
#if defined(USE_ASYNC_COPY)
    stageTile(0, 0);
    __builtin_amdgcn_s_wait_asynccnt(0);
    __syncthreads();
#else
    loadTile(0); storeTile(0);
    __syncthreads();
#endif

    for (int kt = 0; kt < nk; ++kt) {
        const int cur = kt & 1;
#if defined(USE_ASYNC_COPY)
        if (kt + 1 < nk) stageTile((kt + 1) * GBK, cur ^ 1);   // DMA overlaps WMMA
#else
        if (kt + 1 < nk) loadTile((kt + 1) * GBK);             // loads overlap WMMA
#endif

        bf16x16 afrag[4], bfrag[2];
        #pragma unroll
        for (int mi = 0; mi < 4; ++mi) {
            const bf16_t* ap = &sA[cur][(waveM * 64 + mi * 16 + lrow) * LDK];
            union { bf16x16 v; bf16x8 h[2]; } u;
            u.h[0] = *reinterpret_cast<const bf16x8*>(ap + lhi * 8);        // K = lhi*8..+7
            u.h[1] = *reinterpret_cast<const bf16x8*>(ap + 16 + lhi * 8);   // K = 16+lhi*8..+7
            afrag[mi] = u.v;
        }
        #pragma unroll
        for (int ni = 0; ni < 2; ++ni) {
            const bf16_t* bp = &sB[cur][(waveN * 32 + ni * 16 + lrow) * LDK + lhi * 8];
            union { bf16x16 v; bf16x8 h[2]; } u;
            u.h[0] = *reinterpret_cast<const bf16x8*>(bp);
            u.h[1] = *reinterpret_cast<const bf16x8*>(bp + 16);
            bfrag[ni] = u.v;
        }
        #pragma unroll
        for (int mi = 0; mi < 4; ++mi)
            #pragma unroll
            for (int ni = 0; ni < 2; ++ni)
                acc[mi][ni] = __builtin_amdgcn_wmma_f32_16x16x32_bf16(
                    false, afrag[mi], false, bfrag[ni],
                    (short)0, acc[mi][ni], false, false);

#if defined(USE_ASYNC_COPY)
        if (kt + 1 < nk) __builtin_amdgcn_s_wait_asynccnt(0);
        __syncthreads();
#else
        if (kt + 1 < nk) storeTile(cur ^ 1);  // prev readers of buf cur^1 crossed last barrier
        __syncthreads();
#endif
    }

    // epilogue (C layout: VGPR r -> M = r + lhi*8, N = lrow)
    #pragma unroll
    for (int mi = 0; mi < 4; ++mi) {
        #pragma unroll
        for (int ni = 0; ni < 2; ++ni) {
            #pragma unroll
            for (int r = 0; r < 8; ++r) {
                int row = rowBase + waveM * 64 + mi * 16 + r + lhi * 8;
                int col = colBase + waveN * 32 + ni * 16 + lrow;
                size_t idx = (size_t)row * N + col;
                float a = acc[mi][ni][r];
                if constexpr (MODE == 0) Cf[idx] = a;
                else if constexpr (MODE == 1) Cf[idx] = a + Res[idx];
                else if constexpr (MODE == 3) Cb[idx] = (bf16_t)siluf_(a);
                else if constexpr (MODE == 4) Cb[idx] = (bf16_t)(a * (float)Cb[idx]);
                else if constexpr (MODE == 5) Cb[idx] = (bf16_t)sigmoidf_(a + bias[col]);
            }
        }
    }
}

// ---------------------------------------------------------------------------
// gamma = sigmoid(x_conv @ Wg + bg)   [token, head]; one wave per head
// ---------------------------------------------------------------------------
__global__ void gamma_kernel(const bf16_t* __restrict__ xconv,
                             const float* __restrict__ Wg, const float* __restrict__ bg,
                             float* __restrict__ gamma) {
    size_t tok = blockIdx.x;
    int h = threadIdx.x >> 5;
    int lane = threadIdx.x & 31;
    const bf16_t* xr = xconv + tok * DM;
    float acc = 0.f;
    for (int c = lane; c < DM; c += 32) acc += (float)xr[c] * Wg[c * NH + h];
    #pragma unroll
    for (int m = 16; m >= 1; m >>= 1) acc += __shfl_xor(acc, m, 32);
    if (lane == 0) gamma[tok * NH + h] = sigmoidf_(acc + bg[h]);
}

// ---------------------------------------------------------------------------
// per (token, head): q <- l2norm(q); v <- LN(v); u = sigmoid_ig * l2norm(k) * v
// ---------------------------------------------------------------------------
__global__ void qkvu_kernel(float* __restrict__ q, const float* __restrict__ k,
                            float* __restrict__ uv, const bf16_t* __restrict__ ig,
                            const float* __restrict__ vnw, const float* __restrict__ vnb) {
    size_t th = blockIdx.x;                    // token*NH + h
    int d = threadIdx.x;
    size_t base = th * HD + d;
    float qv = q[base], kv = k[base], vv = uv[base];
    __shared__ float red[128];
    float qs  = blockReduce128(qv * qv, red);
    float ks  = blockReduce128(kv * kv, red);
    float vs  = blockReduce128(vv, red);
    float vss = blockReduce128(vv * vv, red);
    float qn = qv * rsqrtf(qs + 1e-12f);
    float kn = kv * rsqrtf(ks + 1e-12f);
    float vm = vs * (1.f / HD);
    float vvar = vss * (1.f / HD) - vm * vm;
    float vn = (vv - vm) * rsqrtf(vvar + EPSF) * vnw[d] + vnb[d];
    q[base]  = qn;
    uv[base] = (float)ig[base] * kn * vn;
}

// ---------------------------------------------------------------------------
// chunked gated scan: mem_t = g_t * mem_{t-1} + u_t  (g per (t,h), d-invariant)
// ---------------------------------------------------------------------------
__global__ void scan_phaseA(float* __restrict__ u, const float* __restrict__ gamma,
                            float* __restrict__ Pbuf) {
    int ck = blockIdx.x % NCH;
    int bh = blockIdx.x / NCH;            // b*NH + h
    int h = bh % NH, b = bh / NH;
    int d = threadIdx.x;
    float m = 0.f, p = 1.f;
    int t0 = ck * LC;
    for (int i = 0; i < LC; ++i) {
        size_t tok = (size_t)b * TSEQ + t0 + i;
        float g = gamma[tok * NH + h];
        size_t idx = (tok * NH + h) * (size_t)HD + d;
        m = g * m + u[idx];
        u[idx] = m;
        p *= g;
        if (d == 0) Pbuf[tok * NH + h] = p;   // prod of g over [t0..t]
    }
}

__global__ void scan_phaseB(const float* __restrict__ u, const float* __restrict__ Pbuf,
                            float* __restrict__ carry) {
    int bh = blockIdx.x;                  // b*NH + h
    int h = bh % NH, b = bh / NH;
    int d = threadIdx.x;
    float m = 0.f;
    for (int ck = 0; ck < NCH; ++ck) {
        carry[((size_t)bh * NCH + ck) * HD + d] = m;   // incoming state of chunk ck
        size_t tok = (size_t)b * TSEQ + ck * LC + LC - 1;
        float G = Pbuf[tok * NH + h];                  // full-chunk gamma product
        float mlast = u[(tok * NH + h) * (size_t)HD + d];
        m = G * m + mlast;
    }
}

__global__ void scan_phaseC(float* __restrict__ u, const float* __restrict__ Pbuf,
                            const float* __restrict__ carry) {
    size_t idx = (size_t)blockIdx.x * 256 + threadIdx.x;   // over MTOK*DM
    int d = (int)(idx % HD);
    size_t th = idx / HD;                 // tok*NH + h
    int h = (int)(th % NH);
    size_t tok = th / NH;
    int t = (int)(tok % TSEQ);
    int b = (int)(tok / TSEQ);
    int ck = t / LC;
    float c = carry[(((size_t)(b * NH + h)) * NCH + ck) * HD + d];
    u[idx] += c * Pbuf[th];
}

// ---------------------------------------------------------------------------
// per (token, head): out = GroupNorm( LN(mem)*q ) * sigmoid_og  -> bf16
// ---------------------------------------------------------------------------
__global__ void post_kernel(const float* __restrict__ mem, const float* __restrict__ q,
                            const bf16_t* __restrict__ og,
                            const float* __restrict__ mnw, const float* __restrict__ mnb,
                            const float* __restrict__ gnw, const float* __restrict__ gnb,
                            bf16_t* __restrict__ out) {
    size_t th = blockIdx.x;
    int d = threadIdx.x;
    int h = (int)(th % NH);
    size_t base = th * HD + d;
    float mv = mem[base];
    __shared__ float red[128];
    float s  = blockReduce128(mv, red);
    float ss = blockReduce128(mv * mv, red);
    float mean = s * (1.f / HD), var = ss * (1.f / HD) - mean * mean;
    float mn = (mv - mean) * rsqrtf(var + EPSF) * mnw[d] + mnb[d];
    float o = mn * q[base];
    float s2  = blockReduce128(o, red);
    float ss2 = blockReduce128(o * o, red);
    float m2 = s2 * (1.f / HD), v2 = ss2 * (1.f / HD) - m2 * m2;
    float gv = (o - m2) * rsqrtf(v2 + EPSF) * gnw[h * HD + d] + gnb[h * HD + d];
    out[base] = (bf16_t)(gv * (float)og[base]);
}

// ---------------------------------------------------------------------------
// host-side orchestration
// ---------------------------------------------------------------------------
extern "C" void kernel_launch(void* const* d_in, const int* in_sizes, int n_in,
                              void* d_out, int out_size, void* d_ws, size_t ws_size,
                              hipStream_t stream) {
    const float* x     = (const float*)d_in[0];
    const float* Wq    = (const float*)d_in[1];
    const float* Wk    = (const float*)d_in[2];
    const float* Wv    = (const float*)d_in[3];
    const float* Wo    = (const float*)d_in[4];
    const float* convw = (const float*)d_in[5];
    const float* convb = (const float*)d_in[6];
    const float* Wig   = (const float*)d_in[7];
    const float* big   = (const float*)d_in[8];
    const float* Wog   = (const float*)d_in[9];
    const float* bog   = (const float*)d_in[10];
    const float* Wg    = (const float*)d_in[11];
    const float* bg    = (const float*)d_in[12];
    const float* vnw   = (const float*)d_in[13];
    const float* vnb   = (const float*)d_in[14];
    const float* mnw   = (const float*)d_in[15];
    const float* mnb   = (const float*)d_in[16];
    const float* gnw   = (const float*)d_in[17];
    const float* gnb   = (const float*)d_in[18];
    const float* ln1w  = (const float*)d_in[19];
    const float* ln1b  = (const float*)d_in[20];
    const float* ln2w  = (const float*)d_in[21];
    const float* ln2b  = (const float*)d_in[22];
    const float* Wgate = (const float*)d_in[23];
    const float* Wval  = (const float*)d_in[24];
    const float* Wout  = (const float*)d_in[25];

    char* ws = (char*)d_ws;
    size_t off = 0;
    auto alloc = [&](size_t bytes) -> void* {
        void* p = ws + off;
        off += (bytes + 255) & ~(size_t)255;
        return p;
    };

    const size_t MC = (size_t)MTOK * DM;       // 16.7M elems
    const size_t MF = (size_t)MTOK * FF;       // 67.1M elems

    float*  h32   = (float*) alloc(MC * 4);
    bf16_t* hb    = (bf16_t*)alloc(MC * 2);
    bf16_t* xcb   = (bf16_t*)alloc(MC * 2);
    float*  qlin  = (float*) alloc(MC * 4);
    float*  klin  = (float*) alloc(MC * 4);
    float*  uvbuf = (float*) alloc(MC * 4);    // vlin -> u -> mem (in place)
    bf16_t* igb   = (bf16_t*)alloc(MC * 2);
    bf16_t* ogb   = (bf16_t*)alloc(MC * 2);
    float*  gamma = (float*) alloc((size_t)MTOK * NH * 4);
    float*  Pbuf  = (float*) alloc((size_t)MTOK * NH * 4);
    float*  carry = (float*) alloc((size_t)NB * NH * NCH * HD * 4);
    bf16_t* outb  = (bf16_t*)alloc(MC * 2);
    float*  h1    = (float*) alloc(MC * 4);
    bf16_t* h2b   = (bf16_t*)alloc(MC * 2);
    bf16_t* mlpb  = (bf16_t*)alloc(MF * 2);
    bf16_t* wqb   = (bf16_t*)alloc((size_t)DM * DM * 2);   // all weights stored [N,K]
    bf16_t* wkb   = (bf16_t*)alloc((size_t)DM * DM * 2);
    bf16_t* wvb   = (bf16_t*)alloc((size_t)DM * DM * 2);
    bf16_t* wob   = (bf16_t*)alloc((size_t)DM * DM * 2);
    bf16_t* wigb  = (bf16_t*)alloc((size_t)DM * DM * 2);
    bf16_t* wogb  = (bf16_t*)alloc((size_t)DM * DM * 2);
    bf16_t* wgtb  = (bf16_t*)alloc((size_t)DM * FF * 2);
    bf16_t* wvlb  = (bf16_t*)alloc((size_t)DM * FF * 2);
    bf16_t* woutb = (bf16_t*)alloc((size_t)FF * DM * 2);
    (void)ws_size; (void)in_sizes; (void)n_in; (void)out_size;

    // cast + transpose weights: in [K,N] f32 -> out [N,K] bf16
    auto castT = [&](const float* src, bf16_t* dst, int K, int N) {
        transpose_cast_kernel<<<dim3(N / 32, K / 32), 256, 0, stream>>>(src, dst, K, N);
    };
    castT(Wq,  wqb,  DM, DM);
    castT(Wk,  wkb,  DM, DM);
    castT(Wv,  wvb,  DM, DM);
    castT(Wo,  wob,  DM, DM);
    castT(Wig, wigb, DM, DM);
    castT(Wog, wogb, DM, DM);
    castT(Wgate, wgtb, DM, FF);
    castT(Wval,  wvlb, DM, FF);
    castT(Wout,  woutb, FF, DM);

    // ln1 -> h (f32 + bf16)
    ln_kernel<<<MTOK, 256, 0, stream>>>(x, ln1w, ln1b, h32, hb);
    // depthwise causal conv + silu -> x_conv (bf16)
    conv_silu_kernel<<<(unsigned)(MC / 256), 256, 0, stream>>>(h32, convw, convb, xcb);

    const dim3 gemB(256);
    dim3 gCC(DM / GBN, MTOK / GBM);     // N=2048
    dim3 gCF(FF / GBN, MTOK / GBM);     // N=8192

    // projections (f32 accum out for downstream norms)
    gemm_bf16_kernel<0><<<gCC, gemB, 0, stream>>>(hb, wqb, qlin, nullptr, nullptr, nullptr, MTOK, DM, DM);
    gemm_bf16_kernel<0><<<gCC, gemB, 0, stream>>>(hb, wkb, klin, nullptr, nullptr, nullptr, MTOK, DM, DM);
    gemm_bf16_kernel<0><<<gCC, gemB, 0, stream>>>(hb, wvb, uvbuf, nullptr, nullptr, nullptr, MTOK, DM, DM);
    // gates fused with sigmoid(+bias)
    gemm_bf16_kernel<5><<<gCC, gemB, 0, stream>>>(xcb, wigb, nullptr, igb, nullptr, big, MTOK, DM, DM);
    gemm_bf16_kernel<5><<<gCC, gemB, 0, stream>>>(xcb, wogb, nullptr, ogb, nullptr, bog, MTOK, DM, DM);
    // per-head decay gates
    gamma_kernel<<<MTOK, 512, 0, stream>>>(xcb, Wg, bg, gamma);

    // per-head normalizations + u
    qkvu_kernel<<<MTOK * NH, 128, 0, stream>>>(qlin, klin, uvbuf, igb, vnw, vnb);

    // chunked gated scan (u -> mem in place)
    scan_phaseA<<<NB * NH * NCH, 128, 0, stream>>>(uvbuf, gamma, Pbuf);
    scan_phaseB<<<NB * NH, 128, 0, stream>>>(uvbuf, Pbuf, carry);
    scan_phaseC<<<(unsigned)(MC / 256), 256, 0, stream>>>(uvbuf, Pbuf, carry);

    // LN(mem)*q -> GroupNorm -> *sigmoid_og -> bf16
    post_kernel<<<MTOK * NH, 128, 0, stream>>>(uvbuf, qlin, ogb, mnw, mnb, gnw, gnb, outb);

    // h1 = x + out @ Wo
    gemm_bf16_kernel<1><<<gCC, gemB, 0, stream>>>(outb, wob, h1, nullptr, x, nullptr, MTOK, DM, DM);

    // ln2 -> h2 (bf16 only)
    ln_kernel<<<MTOK, 256, 0, stream>>>(h1, ln2w, ln2b, nullptr, h2b);

    // SwiGLU MLP: mlpb = silu(h2@Wgate); mlpb *= h2@Wval; out = h1 + mlpb@Wout
    gemm_bf16_kernel<3><<<gCF, gemB, 0, stream>>>(h2b, wgtb, nullptr, mlpb, nullptr, nullptr, MTOK, FF, DM);
    gemm_bf16_kernel<4><<<gCF, gemB, 0, stream>>>(h2b, wvlb, nullptr, mlpb, nullptr, nullptr, MTOK, FF, DM);
    gemm_bf16_kernel<1><<<gCC, gemB, 0, stream>>>(mlpb, woutb, (float*)d_out, nullptr, h1, nullptr, MTOK, DM, FF);
}
